// SelectiveStateSpace_16509854286034
// MI455X (gfx1250) — matrix operand order repaired and checked by reading, not compiled, not run on hardware
//
#include <hip/hip_runtime.h>
#include <cstdint>
#include <cstddef>

// ---------------------------------------------------------------------------
// Problem dimensions (fixed by the reference)
// ---------------------------------------------------------------------------
#define LQ       2048
#define DMODEL   1024
#define DINNER   1024
#define DTRANK   64
#define DSTATE   16
#define PROJ     (DTRANK + 2 * DINNER * DSTATE)   // 32832

typedef __bf16 bf16_t;
typedef __attribute__((ext_vector_type(16))) __bf16 v16bf;
typedef __attribute__((ext_vector_type(8)))  __bf16 v8bf;
typedef __attribute__((ext_vector_type(8)))  float  v8f;
typedef __attribute__((ext_vector_type(4)))  int    v4i;

// ---------------------------------------------------------------------------
// gfx1250 async global->LDS path (guarded; falls back to plain LDS staging)
// ---------------------------------------------------------------------------
#if defined(__AMDGCN__) && \
    __has_builtin(__builtin_amdgcn_global_load_async_to_lds_b128) && \
    __has_builtin(__builtin_amdgcn_s_wait_asynccnt)
#define HAS_ASYNC_LDS 1
#else
#define HAS_ASYNC_LDS 0
#endif

__device__ __forceinline__ void stage16(bf16_t* lds_dst, const bf16_t* gsrc) {
#if HAS_ASYNC_LDS
  __builtin_amdgcn_global_load_async_to_lds_b128(
      (__attribute__((address_space(1))) v4i*)(gsrc),
      (__attribute__((address_space(3))) v4i*)(lds_dst),
      /*offset=*/0, /*cpol=*/0);
#else
  *(uint4*)lds_dst = *(const uint4*)gsrc;
#endif
}

#if HAS_ASYNC_LDS
#define ASYNC_WAIT(n) __builtin_amdgcn_s_wait_asynccnt(n)
#else
#define ASYNC_WAIT(n)
#endif

// ---------------------------------------------------------------------------
// fp32 -> bf16 conversion (grid-stride)
// ---------------------------------------------------------------------------
__global__ void f32_to_bf16_kernel(const float* __restrict__ in,
                                   bf16_t* __restrict__ out, int n) {
  int i = blockIdx.x * blockDim.x + threadIdx.x;
  int stride = gridDim.x * blockDim.x;
  for (; i < n; i += stride) out[i] = (bf16_t)in[i];
}

// ---------------------------------------------------------------------------
// WMMA GEMM: C[M,N] (fp32) = A[M,K] (bf16 row-major) * B[N,K]^T (bf16 row-
// major, K-contiguous).  Block tile 128x128, K-step 64, double-buffered LDS
// with async global->LDS staging.  8 waves; each wave owns a 64x32 output
// tile -> 4x2 accumulators, 16 v_wmma_f32_16x16x32_bf16 per K iteration.
// ---------------------------------------------------------------------------
#define BM 128
#define BN 128
#define BK 64
#define KP 72           // 64 + 8 pad: 144B row pitch, 16B aligned, no bank dup
#define STAGE_OPS 8     // async b128 ops issued per thread per tile stage

__device__ __forceinline__ v16bf load_a_frag(const bf16_t* row, int lane_hi) {
  // 16-bit A 16x32 layout: lane holds K = [8*hi .. 8*hi+7] and [+16 .. +23]
  int k0 = lane_hi * 8;
  v8bf c0 = *(const v8bf*)(row + k0);
  v8bf c1 = *(const v8bf*)(row + k0 + 16);
  v16bf f;
#pragma unroll
  for (int e = 0; e < 8; ++e) { f[e] = c0[e]; f[e + 8] = c1[e]; }
  return f;
}

__device__ __forceinline__ v16bf load_b_frag(const bf16_t* row, int lane_hi) {
  // 16-bit B 32x16 layout: lane holds K = [8*hi .. 8*hi+7] and [+16 .. +23]
  int k0 = lane_hi * 8;
  v8bf c0 = *(const v8bf*)(row + k0);
  v8bf c1 = *(const v8bf*)(row + k0 + 16);
  v16bf f;
#pragma unroll
  for (int e = 0; e < 8; ++e) { f[e] = c0[e]; f[e + 8] = c1[e]; }
  return f;
}

__global__ __launch_bounds__(256)
void gemm_bf16_nt_kernel(const bf16_t* __restrict__ A,
                         const bf16_t* __restrict__ B,
                         float* __restrict__ C, int M, int N, int K) {
  __shared__ bf16_t sa[2][BM * KP];
  __shared__ bf16_t sb[2][BN * KP];

  const int tid     = threadIdx.x;
  const int lane    = tid & 31;
  const int wid     = tid >> 5;
  const int wm      = (wid & 1) * 64;     // wave row offset (0 / 64)
  const int wn      = (wid >> 1) * 32;    // wave col offset (0/32/64/96)
  const int lane_hi = lane >> 4;
  const int lane_lo = lane & 15;
  const int m0      = blockIdx.y * BM;
  const int n0      = blockIdx.x * BN;

  // stage one 128x64 A tile + 128x64 B tile into LDS buffer `buf`
  auto stage = [&](int buf, int kt) {
#pragma unroll
    for (int i = 0; i < 4; ++i) {          // A: 1024 x 16B chunks, 4/thread
      int idx = tid + i * 256;
      int r = idx >> 3, c = idx & 7;
      stage16(&sa[buf][r * KP + c * 8],
              A + (size_t)(m0 + r) * K + kt + c * 8);
    }
#pragma unroll
    for (int i = 0; i < 4; ++i) {          // B: 1024 x 16B chunks, 4/thread
      int idx = tid + i * 256;
      int r = idx >> 3, c = idx & 7;
      int rg = n0 + r;
      if (rg > N - 1) rg = N - 1;          // clamp (stores are guarded)
      stage16(&sb[buf][r * KP + c * 8],
              B + (size_t)rg * K + kt + c * 8);
    }
  };

  v8f acc[4][2] = {};

  const int nk = K / BK;
  stage(0, 0);
  int cur = 0;

  for (int it = 0; it < nk; ++it) {
    if (it + 1 < nk) {
      stage(cur ^ 1, (it + 1) * BK);       // prefetch next tile (async)
      ASYNC_WAIT(STAGE_OPS);               // wait only for the CURRENT tile
    } else {
      ASYNC_WAIT(0);                       // drain
    }
    __syncthreads();

    const bf16_t* pa = sa[cur];
    const bf16_t* pb = sb[cur];
#pragma unroll
    for (int h = 0; h < 2; ++h) {          // two 32-wide K slices
      v16bf af[4], bfm[2];
#pragma unroll
      for (int i = 0; i < 4; ++i)
        af[i] = load_a_frag(pa + (wm + i * 16 + lane_lo) * KP + h * 32,
                            lane_hi);
#pragma unroll
      for (int j = 0; j < 2; ++j)
        bfm[j] = load_b_frag(pb + (wn + j * 16 + lane_lo) * KP + h * 32,
                             lane_hi);
#pragma unroll
      for (int i = 0; i < 4; ++i)
#pragma unroll
        for (int j = 0; j < 2; ++j)
          acc[i][j] = __builtin_amdgcn_wmma_f32_16x16x32_bf16(
              false, af[i], false, bfm[j], (short)0, acc[i][j], false, false);
    }
    __syncthreads();                       // all waves done reading buf `cur`
    cur ^= 1;
  }

  // epilogue: 16x16 f32 C layout -> lane_hi selects M+0 / M+8 half
#pragma unroll
  for (int i = 0; i < 4; ++i) {
#pragma unroll
    for (int j = 0; j < 2; ++j) {
      int col = n0 + wn + j * 16 + lane_lo;
      if (col < N) {
        int rbase = m0 + wm + i * 16 + lane_hi * 8;
        float* cp = C + (size_t)rbase * N + col;
#pragma unroll
        for (int v = 0; v < 8; ++v) cp[(size_t)v * N] = acc[i][j][v];
      }
    }
  }
}

// ---------------------------------------------------------------------------
// delta = softplus(dbc[:, :64] @ W_delta^T)  — one block per timestep
// ---------------------------------------------------------------------------
__global__ __launch_bounds__(256)
void delta_kernel(const float* __restrict__ dbc,
                  const float* __restrict__ W_delta,
                  float* __restrict__ delta) {
  __shared__ float dl[DTRANK];
  const int t = blockIdx.x;
  if (threadIdx.x < DTRANK)
    dl[threadIdx.x] = dbc[(size_t)t * PROJ + threadIdx.x];
  __syncthreads();
  for (int d = threadIdx.x; d < DINNER; d += blockDim.x) {
    const float* w = W_delta + (size_t)d * DTRANK;
    float acc = 0.f;
#pragma unroll
    for (int r = 0; r < DTRANK; ++r) acc = __builtin_fmaf(dl[r], w[r], acc);
    float sp = (acc > 20.f) ? acc : log1pf(__expf(acc));
    delta[(size_t)t * DINNER + d] = sp;
  }
}

// ---------------------------------------------------------------------------
// Selective scan. One lane per (channel d, state s); wave32 = 2 channels per
// wave. 16-lane shfl_xor reduction contracts the state dim. Streams the
// 269 MB B/C slab once, prefetching ahead.
// ---------------------------------------------------------------------------
__global__ __launch_bounds__(256)
void scan_kernel(const float* __restrict__ dbc,
                 const float* __restrict__ delta,
                 const float* __restrict__ x,
                 const float* __restrict__ A_log,
                 const float* __restrict__ Dvec,
                 bf16_t* __restrict__ yb) {
  const int s = threadIdx.x & (DSTATE - 1);
  const int d = blockIdx.x * (blockDim.x / DSTATE) + (threadIdx.x >> 4);

  const float Aval = -__expf(A_log[(size_t)d * DSTATE + s]);
  const float Dd   = Dvec[d];

  const float* Bp = dbc + DTRANK + (size_t)d * DSTATE + s;
  const float* Cp = Bp + (size_t)DINNER * DSTATE;

  float h = 0.f;
  for (int t = 0; t < LQ; ++t) {
    const size_t ro = (size_t)t * PROJ;
    if (t + 8 < LQ) {  // global_prefetch_b8 the streaming B/C rows
      __builtin_prefetch(Bp + (size_t)(t + 8) * PROJ, 0, 0);
      __builtin_prefetch(Cp + (size_t)(t + 8) * PROJ, 0, 0);
    }
    const float dt = delta[(size_t)t * DINNER + d];
    const float xv = x[(size_t)t * DINNER + d];
    const float Bv = Bp[ro];
    const float Cv = Cp[ro];

    const float dA = __expf(dt * Aval);
    h = __builtin_fmaf(dA, h, dt * Bv * xv);

    float p = h * Cv;
    p += __shfl_xor(p, 1);
    p += __shfl_xor(p, 2);
    p += __shfl_xor(p, 4);
    p += __shfl_xor(p, 8);
    if (s == 0)
      yb[(size_t)t * DINNER + d] = (bf16_t)(p + xv * Dd);
  }
}

// ---------------------------------------------------------------------------
// Host-side orchestration
// ---------------------------------------------------------------------------
extern "C" void kernel_launch(void* const* d_in, const int* in_sizes, int n_in,
                              void* d_out, int out_size, void* d_ws,
                              size_t ws_size, hipStream_t stream) {
  const float* x       = (const float*)d_in[0];  // (L, D_INNER)
  const float* W_in    = (const float*)d_in[1];  // (PROJ, D_MODEL)
  const float* W_delta = (const float*)d_in[2];  // (D_INNER, DT_RANK)
  const float* A_log   = (const float*)d_in[3];  // (D_INNER, D_STATE)
  const float* Dv      = (const float*)d_in[4];  // (D_INNER,)
  const float* W_out   = (const float*)d_in[5];  // (D_MODEL, D_INNER)
  float* out           = (float*)d_out;          // (L, D_MODEL)

  char* ws = (char*)d_ws;
  size_t off = 0;
  auto alloc = [&](size_t bytes) -> void* {
    void* p = ws + off;
    off = (off + bytes + 255) & ~(size_t)255;
    return p;
  };

  bf16_t* xb    = (bf16_t*)alloc((size_t)LQ * DINNER * sizeof(bf16_t));
  bf16_t* Winb  = (bf16_t*)alloc((size_t)PROJ * DMODEL * sizeof(bf16_t));
  bf16_t* Woutb = (bf16_t*)alloc((size_t)DMODEL * DINNER * sizeof(bf16_t));
  bf16_t* yb    = (bf16_t*)alloc((size_t)LQ * DINNER * sizeof(bf16_t));
  float*  dbc   = (float*)alloc((size_t)LQ * PROJ * sizeof(float));
  float*  delta = (float*)alloc((size_t)LQ * DINNER * sizeof(float));

  // 1) fp32 -> bf16 conversions
  {
    int n1 = LQ * DINNER;
    f32_to_bf16_kernel<<<2048, 256, 0, stream>>>(x, xb, n1);
    int n2 = PROJ * DMODEL;
    f32_to_bf16_kernel<<<8192, 256, 0, stream>>>(W_in, Winb, n2);
    int n3 = DMODEL * DINNER;
    f32_to_bf16_kernel<<<1024, 256, 0, stream>>>(W_out, Woutb, n3);
  }

  // 2) dbc = x @ W_in^T   (M=2048, N=32832, K=1024)
  {
    dim3 grid((PROJ + BN - 1) / BN, LQ / BM);
    gemm_bf16_nt_kernel<<<grid, 256, 0, stream>>>(xb, Winb, dbc, LQ, PROJ,
                                                  DMODEL);
  }

  // 3) delta = softplus(delta_lr @ W_delta^T)
  delta_kernel<<<LQ, 256, 0, stream>>>(dbc, W_delta, delta);

  // 4) selective scan -> y (bf16, includes + x*D)
  scan_kernel<<<DINNER / (256 / DSTATE), 256, 0, stream>>>(dbc, delta, x,
                                                           A_log, Dv, yb);

  // 5) out = y @ W_out^T  (M=2048, N=1024, K=1024)
  {
    dim3 grid((DMODEL + BN - 1) / BN, LQ / BM);
    gemm_bf16_nt_kernel<<<grid, 256, 0, stream>>>(yb, Woutb, out, LQ, DMODEL,
                                                  DINNER);
  }
}
